// SNND2CAFFusion_6356551598299
// MI455X (gfx1250) — matrix operand order repaired and checked by reading, not compiled, not run on hardware
//
#include <hip/hip_runtime.h>
#include <hip/hip_bf16.h>
#include <math.h>

typedef __attribute__((ext_vector_type(16))) _Float16 v16h;
typedef __attribute__((ext_vector_type(8)))  _Float16 v8h;
typedef __attribute__((ext_vector_type(8)))  float    v8f;
typedef __attribute__((ext_vector_type(4)))  float    v4f;

#define DEV __device__ __forceinline__

// ---------------------------------------------------------------------------
// WMMA helpers (gfx1250: v_wmma_f32_16x16x32_f16, wave32)
// ---------------------------------------------------------------------------
DEV v8f wmma16(v16h a, v16h b, v8f c) {
  return __builtin_amdgcn_wmma_f32_16x16x32_f16(false, a, false, b, (short)0, c,
                                                false, false);
}

DEV v16h cat8(v8h lo, v8h hi) {
  return __builtin_shufflevector(lo, hi, 0, 1, 2, 3, 4, 5, 6, 7, 8, 9, 10, 11,
                                 12, 13, 14, 15);
}

// A-matrix 16x32 (MxK) fragment from row-major [m][k] f16 (LDS or global),
// ld multiple of 8. Each lane: two contiguous 8-half runs => 2x b128 loads.
DEV v16h load_a_f16(const _Float16* base, int row0, int col0, int ld) {
  const int lane = threadIdx.x & 31;
  const int m = lane & 15, hh = lane >> 4;
  const _Float16* p = base + (row0 + m) * ld + col0 + hh * 8;
  const v8h lo = *(const v8h*)p;
  const v8h hi = *(const v8h*)(p + 16);
  return cat8(lo, hi);
}

// B-matrix 32x16 (KxN) fragment from TRANSPOSED [n][k] f16 LDS (ld mult of 8).
// lane needs K = h*8..h*8+7 and 16+h*8..16+h*8+7 at fixed n => 2x ds_load_b128.
DEV v16h load_b_ldsT(const _Float16* base, int krow0, int ncol0, int ld) {
  const int lane = threadIdx.x & 31;
  const int n = lane & 15, hh = lane >> 4;
  const _Float16* p = base + (ncol0 + n) * ld + krow0 + hh * 8;
  const v8h lo = *(const v8h*)p;
  const v8h hi = *(const v8h*)(p + 16);
  return cat8(lo, hi);
}

// group-of-16 lane reductions (rows of a C/D fragment span 16 lanes)
DEV float grp16_max(float v) {
#pragma unroll
  for (int m = 1; m < 16; m <<= 1) v = fmaxf(v, __shfl_xor(v, m, 32));
  return v;
}
DEV float grp16_sum(float v) {
#pragma unroll
  for (int m = 1; m < 16; m <<= 1) v += __shfl_xor(v, m, 32);
  return v;
}

DEV v8f zero8() {
  v8f z;
#pragma unroll
  for (int r = 0; r < 8; ++r) z[r] = 0.0f;
  return z;
}

#define LDB 136  // padded [n][k] LDS stride (halfs): mult of 8, bank step 4
#define LDV 40   // padded sV/sP stride

// ---------------------------------------------------------------------------
// Kernel 0: one-shot f32 -> f16 conversion of the six 128x128 weights.
// wb layout: [6][128][128] row-major, order We, Wd, Wq, Wk, Wv, Wf.
// ---------------------------------------------------------------------------
__global__ __launch_bounds__(256) void k_wcvt(
    const float* __restrict__ We, const float* __restrict__ Wd,
    const float* __restrict__ Wq, const float* __restrict__ Wk,
    const float* __restrict__ Wv, const float* __restrict__ Wf,
    _Float16* __restrict__ wb) {
  const int i = blockIdx.x * 256 + threadIdx.x;  // 0 .. 6*16384-1
  const float* srcs[6] = {We, Wd, Wq, Wk, Wv, Wf};
  const int m = i >> 14, r = i & 16383;
  wb[i] = (_Float16)srcs[m][r];
}

// ---------------------------------------------------------------------------
// Kernel 1: projection GEMMs + cosine similarity. grid (16,32), 256 thr.
// ew/df written pixel-major [bt][N][C] with packed v8h stores.
// ---------------------------------------------------------------------------
__global__ __launch_bounds__(256) void k_proj(
    const float* __restrict__ edge, const float* __restrict__ dvs,
    const _Float16* __restrict__ wWe, const float* __restrict__ beg,
    const _Float16* __restrict__ wWd, const float* __restrict__ bdg,
    _Float16* __restrict__ ewbuf, _Float16* __restrict__ dfbuf,
    float* __restrict__ l2buf) {
  const int bt = blockIdx.y;
  const int n0 = blockIdx.x * 64;
  __shared__ _Float16 sE[64 * LDB];  // [n][k] transposed stage
  __shared__ _Float16 sD[64 * LDB];
  __shared__ float sSumE[64], sSumD[64], sSumX[64], sSim[64];

  const int tid = threadIdx.x;
  const float* eg = edge + (size_t)bt * 128 * 1024;
  const float* dg = dvs + (size_t)bt * 128 * 1024;
  for (int i = tid; i < 128 * 64; i += 256) {
    const int c = i >> 6, j = i & 63;
    sE[j * LDB + c] = (_Float16)eg[c * 1024 + n0 + j];
    sD[j * LDB + c] = (_Float16)dg[c * 1024 + n0 + j];
  }
  if (tid < 64) { sSumE[tid] = 0.f; sSumD[tid] = 0.f; sSumX[tid] = 0.f; }
  __syncthreads();

  const int w = tid >> 5;
  const int lane = tid & 31;
  const int hh = lane >> 4, nn = lane & 15;
  const int rbase = 16 * w + hh * 8;

  v8f eacc[4], dacc[4];
#pragma unroll
  for (int t = 0; t < 4; ++t) { eacc[t] = zero8(); dacc[t] = zero8(); }

#pragma unroll
  for (int ks = 0; ks < 4; ++ks) {
    const v16h ae = load_a_f16(wWe, 16 * w, 32 * ks, 128);
    const v16h ad = load_a_f16(wWd, 16 * w, 32 * ks, 128);
    v16h bE[4], bD[4];
#pragma unroll
    for (int t = 0; t < 4; ++t) {
      bE[t] = load_b_ldsT(sE, 32 * ks, 16 * t, LDB);
      bD[t] = load_b_ldsT(sD, 32 * ks, 16 * t, LDB);
    }
#pragma unroll
    for (int t = 0; t < 4; ++t) {
      eacc[t] = wmma16(ae, bE[t], eacc[t]);
      dacc[t] = wmma16(ad, bD[t], dacc[t]);
    }
  }

  float biasE[8], biasD[8];
#pragma unroll
  for (int r = 0; r < 8; ++r) {
    biasE[r] = beg[rbase + r];
    biasD[r] = bdg[rbase + r];
  }

#pragma unroll
  for (int t = 0; t < 4; ++t) {
    float se = 0.f, sd = 0.f, sx = 0.f;
#pragma unroll
    for (int r = 0; r < 8; ++r) {
      eacc[t][r] += biasE[r];
      dacc[t][r] += biasD[r];
      const float e = eacc[t][r], d = dacc[t][r];
      se += e * e; sd += d * d; sx += e * d;
    }
    atomicAdd(&sSumE[t * 16 + nn], se);
    atomicAdd(&sSumD[t * 16 + nn], sd);
    atomicAdd(&sSumX[t * 16 + nn], sx);
  }
  __syncthreads();

  if (tid < 64) {
    const float l2e = sqrtf(sSumE[tid]);
    const float l2d = sqrtf(sSumD[tid]);
    sSim[tid] = (sSumX[tid] / ((l2e + 1e-6f) * (l2d + 1e-6f)) + 1.f) * 0.5f;
    l2buf[bt * 1024 + n0 + tid] = l2e;
  }
  __syncthreads();

  _Float16* ewp = ewbuf + (size_t)bt * 1024 * 128;
  _Float16* dfp = dfbuf + (size_t)bt * 1024 * 128;
#pragma unroll
  for (int t = 0; t < 4; ++t) {
    const float sim = sSim[t * 16 + nn];
    const int n = n0 + t * 16 + nn;
    v8h pe, pd;
#pragma unroll
    for (int r = 0; r < 8; ++r) {
      pe[r] = (_Float16)(eacc[t][r] * sim);
      pd[r] = (_Float16)(dacc[t][r]);
    }
    *(v8h*)(ewp + (size_t)n * 128 + rbase) = pe;
    *(v8h*)(dfp + (size_t)n * 128 + rbase) = pd;
  }
}

// ---------------------------------------------------------------------------
// Kernel 2: density = spatial softmax(l2e) * N per bt.
// ---------------------------------------------------------------------------
__global__ __launch_bounds__(256) void k_dens(const float* __restrict__ l2buf,
                                              float* __restrict__ densbuf) {
  const int bt = blockIdx.x;
  __shared__ float sv[1024];
  __shared__ float red[256];
  const float* src = l2buf + bt * 1024;
  const int tid = threadIdx.x;
  float lmax = -1e30f;
  for (int i = tid; i < 1024; i += 256) {
    const float x = src[i];
    sv[i] = x;
    lmax = fmaxf(lmax, x);
  }
  red[tid] = lmax;
  __syncthreads();
  for (int s = 128; s > 0; s >>= 1) {
    if (tid < s) red[tid] = fmaxf(red[tid], red[tid + s]);
    __syncthreads();
  }
  const float mx = red[0];
  __syncthreads();
  float lsum = 0.f;
  for (int i = tid; i < 1024; i += 256) lsum += __expf(sv[i] - mx);
  red[tid] = lsum;
  __syncthreads();
  for (int s = 128; s > 0; s >>= 1) {
    if (tid < s) red[tid] += red[tid + s];
    __syncthreads();
  }
  const float inv = 1024.f / red[0];
  for (int i = tid; i < 1024; i += 256)
    densbuf[bt * 1024 + i] = __expf(sv[i] - mx) * inv;
}

// ---------------------------------------------------------------------------
// Kernel 3: fused Q/K/V GEMMs. q,k,v written pixel-major [bt][N][C], packed.
// ---------------------------------------------------------------------------
__global__ __launch_bounds__(256) void k_qkv(
    const _Float16* __restrict__ ewbuf, const _Float16* __restrict__ dfbuf,
    const float* __restrict__ densbuf, const _Float16* __restrict__ wWq,
    const float* __restrict__ bqg, const _Float16* __restrict__ wWk,
    const float* __restrict__ bkg, const _Float16* __restrict__ wWv,
    const float* __restrict__ bvg, _Float16* __restrict__ qbuf,
    _Float16* __restrict__ kbuf, _Float16* __restrict__ vbuf) {
  const int bt = blockIdx.y;
  const int n0 = blockIdx.x * 64;
  __shared__ _Float16 sEw[64 * LDB];  // [n][k]
  __shared__ _Float16 sDw[64 * LDB];

  const int tid = threadIdx.x;
  const _Float16* ewp = ewbuf + (size_t)bt * 1024 * 128;
  const _Float16* dfp = dfbuf + (size_t)bt * 1024 * 128;
  const float* dnp = densbuf + bt * 1024;
  for (int i = tid; i < 64 * 128; i += 256) {
    const int j = i >> 7, c = i & 127;
    sEw[j * LDB + c] = ewp[(size_t)(n0 + j) * 128 + c];
    sDw[j * LDB + c] =
        (_Float16)((float)dfp[(size_t)(n0 + j) * 128 + c] * dnp[n0 + j]);
  }
  __syncthreads();

  const int w = tid >> 5;
  const int lane = tid & 31;
  const int hh = lane >> 4, nn = lane & 15;
  const int rbase = 16 * w + hh * 8;

  v8f qacc[4], kacc[4], vacc[4];
#pragma unroll
  for (int t = 0; t < 4; ++t) { qacc[t] = zero8(); kacc[t] = zero8(); vacc[t] = zero8(); }

#pragma unroll
  for (int ks = 0; ks < 4; ++ks) {
    const v16h aq = load_a_f16(wWq, 16 * w, 32 * ks, 128);
    const v16h ak = load_a_f16(wWk, 16 * w, 32 * ks, 128);
    const v16h av = load_a_f16(wWv, 16 * w, 32 * ks, 128);
    v16h bE[4], bD[4];
#pragma unroll
    for (int t = 0; t < 4; ++t) {
      bE[t] = load_b_ldsT(sEw, 32 * ks, 16 * t, LDB);
      bD[t] = load_b_ldsT(sDw, 32 * ks, 16 * t, LDB);
    }
#pragma unroll
    for (int t = 0; t < 4; ++t) {
      qacc[t] = wmma16(aq, bE[t], qacc[t]);
      kacc[t] = wmma16(ak, bD[t], kacc[t]);
      vacc[t] = wmma16(av, bD[t], vacc[t]);
    }
  }

  float biasQ[8], biasK[8], biasV[8];
#pragma unroll
  for (int r = 0; r < 8; ++r) {
    biasQ[r] = bqg[rbase + r];
    biasK[r] = bkg[rbase + r];
    biasV[r] = bvg[rbase + r];
  }

  _Float16* qp = qbuf + (size_t)bt * 1024 * 128;
  _Float16* kp = kbuf + (size_t)bt * 1024 * 128;
  _Float16* vp = vbuf + (size_t)bt * 1024 * 128;
#pragma unroll
  for (int t = 0; t < 4; ++t) {
    const int n = n0 + t * 16 + nn;
    v8h pq, pk, pv;
#pragma unroll
    for (int r = 0; r < 8; ++r) {
      pq[r] = (_Float16)(qacc[t][r] + biasQ[r]);
      pk[r] = (_Float16)(kacc[t][r] + biasK[r]);
      pv[r] = (_Float16)(vacc[t][r] + biasV[r]);
    }
    *(v8h*)(qp + (size_t)n * 128 + rbase) = pq;
    *(v8h*)(kp + (size_t)n * 128 + rbase) = pk;
    *(v8h*)(vp + (size_t)n * 128 + rbase) = pv;
  }
}

// ---------------------------------------------------------------------------
// Kernel 4: windowed flash attention. grid (8, 32). Per wave: 16 query rows.
// obuf written channel-major [bt][C][N] with packed v8h stores.
// ---------------------------------------------------------------------------
__global__ __launch_bounds__(256) void k_attn(
    const _Float16* __restrict__ qbuf, const _Float16* __restrict__ kbuf,
    const _Float16* __restrict__ vbuf, _Float16* __restrict__ obuf) {
  const int bt = blockIdx.y;
  const int q0 = blockIdx.x * 128;
  __shared__ _Float16 sQ[128 * LDB];    // [local qrow][c]  (A operand)
  __shared__ _Float16 sK[32 * LDB];     // [key][c]         (B operand for S)
  __shared__ _Float16 sV[128 * LDV];    // [c][key]         (B operand for O)
  __shared__ _Float16 sP[8][16 * LDV];  // per-wave P stage (A operand)

  const int tid = threadIdx.x;
  const _Float16* qg = qbuf + (size_t)bt * 1024 * 128;
  const _Float16* kg = kbuf + (size_t)bt * 1024 * 128;
  const _Float16* vg = vbuf + (size_t)bt * 1024 * 128;
  for (int i = tid; i < 128 * 128; i += 256) {
    const int qr = i >> 7, c = i & 127;
    sQ[qr * LDB + c] = qg[(size_t)(q0 + qr) * 128 + c];
  }

  const int w = tid >> 5;
  const int lane = tid & 31;
  const int hh = lane >> 4, nn = lane & 15;
  const float scale = 0.08838834764831845f;  // 1/sqrt(128)

  v8f o[8];
#pragma unroll
  for (int t = 0; t < 8; ++t) o[t] = zero8();
  float m_r[8], l_r[8];
#pragma unroll
  for (int r = 0; r < 8; ++r) { m_r[r] = -1e30f; l_r[r] = 0.f; }

  for (int kc = 0; kc < 32; ++kc) {
    const int k0 = kc * 32;
    __syncthreads();
    for (int i = tid; i < 32 * 128; i += 256) {
      const int j = i >> 7, c = i & 127;
      sK[j * LDB + c] = kg[(size_t)(k0 + j) * 128 + c];
      sV[c * LDV + j] = vg[(size_t)(k0 + j) * 128 + c];
      if (kc + 1 < 32) {
        __builtin_prefetch(&kg[(size_t)(k0 + 32 + j) * 128 + c], 0, 0);
        __builtin_prefetch(&vg[(size_t)(k0 + 32 + j) * 128 + c], 0, 0);
      }
    }
    __syncthreads();

    // S = Q(16x128) x K^T(128x32): hoist all fragments, then WMMA chain
    v16h aq[4], bk0[4], bk1[4];
#pragma unroll
    for (int ks = 0; ks < 4; ++ks) {
      aq[ks] = load_a_f16(sQ, 16 * w, 32 * ks, LDB);
      bk0[ks] = load_b_ldsT(sK, 32 * ks, 0, LDB);
      bk1[ks] = load_b_ldsT(sK, 32 * ks, 16, LDB);
    }
    v8f s0 = zero8(), s1 = zero8();
#pragma unroll
    for (int ks = 0; ks < 4; ++ks) {
      s0 = wmma16(aq[ks], bk0[ks], s0);
      s1 = wmma16(aq[ks], bk1[ks], s1);
    }

    const int nk0 = k0 + nn, nk1 = k0 + 16 + nn;
    const int kx0 = nk0 >> 5, ky0 = nk0 & 31;
    const int kx1 = nk1 >> 5, ky1 = nk1 & 31;
#pragma unroll
    for (int r = 0; r < 8; ++r) {
      const int qrow = q0 + 16 * w + hh * 8 + r;
      const int qx = qrow >> 5, qy = qrow & 31;
      float v0 = s0[r] * scale;
      float v1 = s1[r] * scale;
      if (abs(kx0 - qx) > 3 || abs(ky0 - qy) > 3) v0 += -10000.f;
      if (abs(kx1 - qx) > 3 || abs(ky1 - qy) > 3) v1 += -10000.f;
      const float mcur = grp16_max(fmaxf(v0, v1));
      const float mnew = fmaxf(m_r[r], mcur);
      const float alpha = __expf(m_r[r] - mnew);
      const float p0 = __expf(v0 - mnew);
      const float p1 = __expf(v1 - mnew);
      const float rs = grp16_sum(p0 + p1);
      l_r[r] = l_r[r] * alpha + rs;
      m_r[r] = mnew;
#pragma unroll
      for (int t = 0; t < 8; ++t) o[t][r] *= alpha;
      const int prow = hh * 8 + r;
      sP[w][prow * LDV + nn] = (_Float16)p0;
      sP[w][prow * LDV + 16 + nn] = (_Float16)p1;
    }
    __syncthreads();

    // O += P(16x32) x V(32x128): hoist fragments, then WMMA chain
    const v16h ap = load_a_f16(sP[w], 0, 0, LDV);
    v16h bv_[8];
#pragma unroll
    for (int t = 0; t < 8; ++t) bv_[t] = load_b_ldsT(sV, 0, 16 * t, LDV);
#pragma unroll
    for (int t = 0; t < 8; ++t) o[t] = wmma16(ap, bv_[t], o[t]);
  }

  float linv[8];
#pragma unroll
  for (int r = 0; r < 8; ++r) linv[r] = 1.f / l_r[r];

  _Float16* op = obuf + (size_t)bt * 128 * 1024;  // [C][N]
  const int qbase = q0 + 16 * w + hh * 8;
#pragma unroll
  for (int t = 0; t < 8; ++t) {
    const int c = t * 16 + nn;
    v8h pk;
#pragma unroll
    for (int r = 0; r < 8; ++r) pk[r] = (_Float16)(o[t][r] * linv[r]);
    *(v8h*)(op + (size_t)c * 1024 + qbase) = pk;
  }
}

// ---------------------------------------------------------------------------
// Kernel 5: fused = Wf * out^T + bf, f32 output [bt][C][N].
// ---------------------------------------------------------------------------
__global__ __launch_bounds__(256) void k_fuse(const _Float16* __restrict__ obuf,
                                              const _Float16* __restrict__ wWf,
                                              const float* __restrict__ bfg,
                                              float* __restrict__ fbuf) {
  const int bt = blockIdx.y;
  const int n0 = blockIdx.x * 64;
  __shared__ _Float16 sO[64 * LDB];  // [n][c] (B operand)

  const int tid = threadIdx.x;
  const _Float16* op = obuf + (size_t)bt * 128 * 1024;  // [C][N]
  for (int i = tid; i < 128 * 64; i += 256) {
    const int c = i >> 6, j = i & 63;
    sO[j * LDB + c] = op[(size_t)c * 1024 + n0 + j];
  }
  __syncthreads();

  const int w = tid >> 5;
  const int lane = tid & 31;
  const int hh = lane >> 4, nn = lane & 15;
  const int rbase = 16 * w + hh * 8;

  v8f acc[4];
#pragma unroll
  for (int t = 0; t < 4; ++t) acc[t] = zero8();
#pragma unroll
  for (int ks = 0; ks < 4; ++ks) {
    const v16h a = load_a_f16(wWf, 16 * w, 32 * ks, 128);
    v16h b[4];
#pragma unroll
    for (int t = 0; t < 4; ++t) b[t] = load_b_ldsT(sO, 32 * ks, 16 * t, LDB);
#pragma unroll
    for (int t = 0; t < 4; ++t) acc[t] = wmma16(a, b[t], acc[t]);
  }

  float biasF[8];
#pragma unroll
  for (int r = 0; r < 8; ++r) biasF[r] = bfg[rbase + r];

  float* fp = fbuf + (size_t)bt * 128 * 1024;
#pragma unroll
  for (int t = 0; t < 4; ++t) {
    const int col = n0 + t * 16 + nn;
#pragma unroll
    for (int r = 0; r < 8; ++r) {
      const int row = rbase + r;
      fp[(size_t)row * 1024 + col] = acc[t][r] + biasF[r];
    }
  }
}

// ---------------------------------------------------------------------------
// Kernel 6: LIF spike scan over T=4 (tau=0.5, thresh=1, hard reset).
// ---------------------------------------------------------------------------
__global__ __launch_bounds__(256) void k_lif(const float* __restrict__ fbuf,
                                             float* __restrict__ out) {
  const size_t i = (size_t)blockIdx.x * 256 + threadIdx.x;
  const size_t total = (size_t)8 * 128 * 1024;  // B*C*H*W
  if (i >= total) return;
  const int n = (int)(i & 1023);
  const int c = (int)((i >> 10) & 127);
  const int b = (int)(i >> 17);
  float mem = 0.f;
#pragma unroll
  for (int t = 0; t < 4; ++t) {
    const size_t off = (((size_t)(b * 4 + t) * 128 + c) << 10) + n;
    mem = mem * 0.5f + fbuf[off];
    const float spk = (mem - 1.0f) > 0.f ? 1.f : 0.f;
    mem = (1.f - spk) * mem;
    out[off] = spk;
  }
}

// ---------------------------------------------------------------------------
extern "C" void kernel_launch(void* const* d_in, const int* in_sizes, int n_in,
                              void* d_out, int out_size, void* d_ws,
                              size_t ws_size, hipStream_t stream) {
  (void)in_sizes; (void)n_in; (void)out_size; (void)ws_size;
  const float* edge = (const float*)d_in[0];
  const float* dvs  = (const float*)d_in[1];
  const float* We = (const float*)d_in[2];
  const float* be = (const float*)d_in[3];
  const float* Wd = (const float*)d_in[4];
  const float* bd = (const float*)d_in[5];
  const float* Wq = (const float*)d_in[6];
  const float* bq = (const float*)d_in[7];
  const float* Wk = (const float*)d_in[8];
  const float* bk = (const float*)d_in[9];
  const float* Wv = (const float*)d_in[10];
  const float* bv = (const float*)d_in[11];
  const float* Wf = (const float*)d_in[12];
  const float* bf = (const float*)d_in[13];
  float* out = (float*)d_out;

  const size_t BT = 32, C = 128, N = 1024;
  const size_t f16b = BT * C * N * sizeof(_Float16);  // 8 MiB
  char* p = (char*)d_ws;
  _Float16* ewbuf = (_Float16*)p; p += f16b;  // [bt][N][C]
  _Float16* dfbuf = (_Float16*)p; p += f16b;  // [bt][N][C]
  _Float16* qbuf  = (_Float16*)p; p += f16b;  // [bt][N][C]
  _Float16* kbuf  = (_Float16*)p; p += f16b;  // [bt][N][C]
  _Float16* vbuf  = (_Float16*)p; p += f16b;  // [bt][N][C]
  _Float16* obuf  = (_Float16*)p; p += f16b;  // [bt][C][N]
  float* l2buf   = (float*)p; p += BT * N * sizeof(float);
  float* densbuf = (float*)p; p += BT * N * sizeof(float);
  float* fbuf    = (float*)p; p += BT * C * N * sizeof(float);
  _Float16* wbuf = (_Float16*)p; p += 6 * C * C * sizeof(_Float16);

  k_wcvt<<<6 * 128 * 128 / 256, 256, 0, stream>>>(We, Wd, Wq, Wk, Wv, Wf, wbuf);
  _Float16* wWe = wbuf + 0 * 16384;
  _Float16* wWd = wbuf + 1 * 16384;
  _Float16* wWq = wbuf + 2 * 16384;
  _Float16* wWk = wbuf + 3 * 16384;
  _Float16* wWv = wbuf + 4 * 16384;
  _Float16* wWf = wbuf + 5 * 16384;

  k_proj<<<dim3(16, 32), 256, 0, stream>>>(edge, dvs, wWe, be, wWd, bd, ewbuf,
                                           dfbuf, l2buf);
  k_dens<<<32, 256, 0, stream>>>(l2buf, densbuf);
  k_qkv<<<dim3(16, 32), 256, 0, stream>>>(ewbuf, dfbuf, densbuf, wWq, bq, wWk,
                                          bk, wWv, bv, qbuf, kbuf, vbuf);
  k_attn<<<dim3(8, 32), 256, 0, stream>>>(qbuf, kbuf, vbuf, obuf);
  k_fuse<<<dim3(16, 32), 256, 0, stream>>>(obuf, wWf, bf, fbuf);
  k_lif<<<(8 * 128 * 1024 + 255) / 256, 256, 0, stream>>>(fbuf, out);
}
